// Cosine_Similarity_86337432584368
// MI455X (gfx1250) — matrix operand order repaired and checked by reading, not compiled, not run on hardware
//
#include <hip/hip_runtime.h>
#include <hip/hip_bf16.h>

// CDNA5 / gfx1250. Cosine-similarity 3x3 neighborhood, H=W=1024, C=128.
// Key identity: sum_{3x3} dot(xn, xn_shift) = dot(xn, boxsum3x3(xn)).
// Memory-bound: ~612 MB traffic -> ~26 us floor at 23.3 TB/s.
// Phase 1 batches 4 independent global_load_b128 per wave iteration for MLP
// (only 2 waves/SIMD at 305 KB LDS, so per-wave outstanding loads carry BW).

typedef __attribute__((ext_vector_type(16))) _Float16 v16h;
typedef __attribute__((ext_vector_type(2)))  _Float16 h2;
typedef __attribute__((ext_vector_type(8)))  float    v8f;

#define H_DIM 1024
#define W_DIM 1024
#define C_DIM 128
#define TILE_W 16
#define TILE_H 32
#define HALO_W (TILE_W + 2)   // 18
#define HALO_H (TILE_H + 2)   // 34
#define XPITCH 132            // padded pixel pitch in halfs (66 dwords -> 16-lane accesses spread banks)
#define EPSV 1e-8f

#define XN_HALFS (HALO_H * HALO_W * XPITCH)   // 80784
#define R_HALFS  (HALO_H * TILE_W * XPITCH)   // 71808
#define SMEM_BYTES ((XN_HALFS + R_HALFS) * 2) // 305184 B < 320 KB/WGP

__global__ void cosine_sim3x3_wmma(const float* __restrict__ x,
                                   float* __restrict__ out) {
    extern __shared__ __align__(16) char smem_raw[];
    _Float16* xn = (_Float16*)smem_raw;           // [34][18][132] f16 normalized
    _Float16* Rr = xn + XN_HALFS;                 // [34][16][132] f16 horizontal 3-sum
    h2* xn2 = (h2*)xn;
    h2* R2  = (h2*)Rr;

    const int tid  = threadIdx.x;
    const int lane = tid & 31;
    const int wv   = tid >> 5;                    // 8 waves
    const int c0   = blockIdx.x * TILE_W;
    const int r0   = blockIdx.y * TILE_H;

    // ---------------- Phase 1: load + normalize halo tile into LDS (f16) ----
    // 612 pixels = 153 groups of 4. Each wave takes a group: issue 4
    // independent b128 loads first, then 4 independent wave-reductions.
    for (int g = wv; g < (HALO_H * HALO_W) / 4; g += 8) {
        float4 v[4];
        const int pbase = g * 4;
        #pragma unroll
        for (int q = 0; q < 4; ++q) {
            const int p   = pbase + q;
            const int row = p / HALO_W;           // 0..33
            const int pix = p - row * HALO_W;     // 0..17
            const int gr  = r0 - 1 + row;
            const int gc  = c0 - 1 + pix;
            v[q] = make_float4(0.f, 0.f, 0.f, 0.f);
            if (gr >= 0 && gr < H_DIM && gc >= 0 && gc < W_DIM) {
                const float4* src =
                    (const float4*)(x + ((size_t)gr * W_DIM + gc) * C_DIM);
                v[q] = src[lane];                 // channels lane*4 .. lane*4+3
            }
        }
        #pragma unroll
        for (int q = 0; q < 4; ++q) {
            float ssq = v[q].x * v[q].x + v[q].y * v[q].y
                      + v[q].z * v[q].z + v[q].w * v[q].w;
            #pragma unroll
            for (int o = 16; o > 0; o >>= 1) ssq += __shfl_xor(ssq, o, 32);
            const float inv = 1.0f / fmaxf(sqrtf(ssq), EPSV);

            h2* dst = (h2*)(xn + (size_t)(pbase + q) * XPITCH + lane * 4);
            h2 a, b;
            a.x = (_Float16)(v[q].x * inv); a.y = (_Float16)(v[q].y * inv);
            b.x = (_Float16)(v[q].z * inv); b.y = (_Float16)(v[q].w * inv);
            dst[0] = a;
            dst[1] = b;
        }
    }
    __syncthreads();

    // ---------------- Phase 2: horizontal 3-sum R (v_pk_add_f16) ------------
    // R[row][wi][c] = xn[row][wi] + xn[row][wi+1] + xn[row][wi+2]
    // 34*16*64 half2 = 34816 -> exactly 136 per thread.
    for (int idx = tid; idx < HALO_H * TILE_W * (C_DIM / 2); idx += 256) {
        const int c2   = idx & 63;
        const int rest = idx >> 6;
        const int wi   = rest & 15;
        const int row  = rest >> 4;
        const int base = (row * HALO_W + wi) * (XPITCH / 2) + c2;
        h2 s = xn2[base] + xn2[base + (XPITCH / 2)] + xn2[base + XPITCH];
        R2[(row * TILE_W + wi) * (XPITCH / 2) + c2] = s;
    }
    __syncthreads();

    // ---------------- Phase 3: out row = diag( Xn_tile * S_tile^T ) ---------
    // S[n] = R[hh] + R[hh+1] + R[hh+2] fused into the B fragment.
    // 4 x v_wmma_f32_16x16x32_f16 accumulate over K=128 per output row.
    const int  m  = lane & 15;
    const bool hi = lane >= 16;

    #pragma unroll
    for (int i = 0; i < 4; ++i) {
        const int hh = wv + 8 * i;                // 0..31 (local output row)
        v8f acc = {};

        #pragma unroll
        for (int kk = 0; kk < 4; ++kk) {          // K slice = channels kk*32..+31
            v16h a, b;
            const int aBase = ((hh + 1) * HALO_W + (m + 1)) * (XPITCH / 2)
                              + kk * 16;          // half2 units
            const int bRow  = hh * TILE_W + m;    // n == lane&15
            const int bBase = bRow * (XPITCH / 2) + kk * 16;
            const int rowStride = TILE_W * (XPITCH / 2);

            #pragma unroll
            for (int pI = 0; pI < 8; ++pI) {
                // A 16x32 f16 layout: K = j + 8*(j>=8) + 8*(lane>=16)
                const int kA = 2 * pI + ((pI >= 4) ? 8 : 0) + (hi ? 8 : 0);
                h2 av = xn2[aBase + (kA >> 1)];
                a[2 * pI]     = av.x;
                a[2 * pI + 1] = av.y;
                // B 32x16 f16 layout: K = j + 8*(j>=8) + 8*(lane>=16); B[k][n] = S[n][k]
                const int kB = 2 * pI + ((pI >= 4) ? 8 : 0) + (hi ? 8 : 0);
                h2 bv = R2[bBase + (kB >> 1)]
                      + R2[bBase + rowStride     + (kB >> 1)]
                      + R2[bBase + 2 * rowStride + (kB >> 1)];
                b[2 * pI]     = bv.x;
                b[2 * pI + 1] = bv.y;
            }
            acc = __builtin_amdgcn_wmma_f32_16x16x32_f16(
                false, a, false, b, (short)0, acc, false, false);
        }

        // Diagonal: lanes 0..7 -> M=lane (VGPR=lane); lanes 24..31 -> M=lane-16
        const int sel = (lane < 16) ? lane : (lane - 24);
        float v = acc[0];
        #pragma unroll
        for (int r = 1; r < 8; ++r) v = (sel == r) ? acc[r] : v;

        if (lane < 8 || lane >= 24) {
            const int M = (lane < 8) ? lane : (lane - 16);
            out[(size_t)(r0 + hh) * W_DIM + (c0 + M)] = v - 1.0f;
        }
    }
}

extern "C" void kernel_launch(void* const* d_in, const int* in_sizes, int n_in,
                              void* d_out, int out_size, void* d_ws, size_t ws_size,
                              hipStream_t stream) {
    (void)in_sizes; (void)n_in; (void)d_ws; (void)ws_size; (void)out_size;
    const float* x  = (const float*)d_in[0];
    float* out      = (float*)d_out;
    dim3 grid(W_DIM / TILE_W, H_DIM / TILE_H);    // 64 x 32 = 2048 workgroups
    dim3 block(256);                              // 8 wave32
    cosine_sim3x3_wmma<<<grid, block, SMEM_BYTES, stream>>>(x, out);
}
